// LIFConvRNN_79242146611234
// MI455X (gfx1250) — matrix operand order repaired and checked by reading, not compiled, not run on hardware
//
#include <hip/hip_runtime.h>

// ---------------------------------------------------------------------------
// LIF Conv-RNN for MI455X (gfx1250).
// Per timestep: fused dual-conv implicit GEMM (K = 576 x + 576 h = 1152) with
// v_wmma_f32_16x16x32_f16. Weights staged into LDS by the Tensor Data Mover
// (tensor_load_to_lds + s_wait_tensorcnt); activation tiles staged by
// cooperative zero-padded vector loads. K-loop fully unrolled; all fragment
// loads are ds_load_b128 with constant offsets off loop-invariant bases, so
// the steady state is pure LDS-load clauses + back-to-back WMMAs.
// ---------------------------------------------------------------------------

typedef _Float16 v8h  __attribute__((ext_vector_type(8)));
typedef _Float16 v16h __attribute__((ext_vector_type(16)));
typedef float    v8f  __attribute__((ext_vector_type(8)));
typedef unsigned int u32x4 __attribute__((ext_vector_type(4)));
typedef int          i32x4 __attribute__((ext_vector_type(4)));
typedef int          i32x8 __attribute__((ext_vector_type(8)));

#define BN    16
#define HH    32
#define WW    32
#define TT    20
#define CH    64
#define KTOT  1152            // (3*3*64) x-part + (3*3*64) h-part
#define WROW  1160            // padded weight row (halfs): 580 dwords % 64 = 4
#define WLDS_HALFS (CH * WROW)          // 74,240 halfs = 148,480 B
#define THRESH 0.5f
#define DECAY  0.3f

// Activation tile: 4 image rows + halo (6) x (32 + halo = 34) cols x 64 ch,
// channel slot padded 64 -> 72 halfs (36 dwords) to spread LDS banks.
#define TROWS 6
#define TCOLS 34
#define PIXP  72
#define TILE_P (TROWS * TCOLS * PIXP)   // 14,688 halfs = 29,376 B

#define DYN_LDS_BYTES ((2 * TILE_P + WLDS_HALFS) * 2)   // 207,232 B

#define SHUF16(lo, hi) __builtin_shufflevector(lo, hi, 0, 1, 2, 3, 4, 5, 6, 7, \
                                               8, 9, 10, 11, 12, 13, 14, 15)

// ---- Pre-pass: data [B,Cin,H,W,T] f32  ->  xT [T][B][H*W][Cin] f16 ----------
__global__ void k_convert_x(const float* __restrict__ data,
                            _Float16* __restrict__ xT, int n) {
    int o = blockIdx.x * 256 + threadIdx.x;
    if (o >= n) return;
    int c = o & 63;
    int tmp = o >> 6;
    int p = tmp & 1023;          // h*32 + w
    tmp >>= 10;
    int b = tmp & 15;
    int t = tmp >> 4;
    float v = data[((size_t)((b * 64 + c) * 1024 + p)) * 20 + t];
    xT[o] = (_Float16)v;
}

// ---- Pre-pass: Wx/Wh [Ch,Cin,3,3] f32 -> Wt [Ch][WROW] f16 (padded rows;
//      k=(ky*3+kx)*64+cin, x-part k<576, h-part 576<=k<1152, pad k>=1152) -----
__global__ void k_convert_w(const float* __restrict__ Wx,
                            const float* __restrict__ Wh,
                            _Float16* __restrict__ Wt, int n) {
    int o = blockIdx.x * 256 + threadIdx.x;
    if (o >= n) return;
    int k  = o % WROW;
    int ch = o / WROW;
    float v = 0.0f;
    if (k < KTOT) {
        const float* src = (k < 576) ? Wx : Wh;
        int kk  = (k < 576) ? k : k - 576;
        int tap = kk >> 6;       // ky*3+kx
        int cin = kk & 63;
        v = src[(ch * 64 + cin) * 9 + tap];
    }
    Wt[o] = (_Float16)v;
}

// ---- Zero a 16B-granular region (u state + both h spike buffers) ------------
__global__ void k_zero16(uint4* __restrict__ p, int n) {
    int i = blockIdx.x * 256 + threadIdx.x;
    if (i < n) { uint4 z = {0u, 0u, 0u, 0u}; p[i] = z; }
}

// ---- One LIF timestep: fused dual-conv implicit GEMM + state update ---------
// 128 blocks x 256 threads (8 wave32). Block = one image's 4-row slab.
// Wave w computes D rows [w*16, w*16+16) x all 64 output channels.
__global__ __launch_bounds__(256)
void k_lif_step(const _Float16* __restrict__ xT,    // [B*1024][64] this t
                const _Float16* __restrict__ hin,   // [B*1024][64] spikes t-1
                const _Float16* __restrict__ Wt,    // [64][WROW] padded
                const float*    __restrict__ bx,
                const float*    __restrict__ bh,
                float*          __restrict__ u,     // [B*1024][64]
                _Float16*       __restrict__ hout,  // [B*1024][64]
                float*          __restrict__ out,   // [B][64][32][32][20]
                int t) {
    extern __shared__ _Float16 smem[];   // [x tile | h tile | weights]
    _Float16* wlds = smem + 2 * TILE_P;

    const int tid  = threadIdx.x;
    const int wave = tid >> 5;
    const int lane = tid & 31;
    const int hl   = lane >> 4;                     // lane-half (0/1)
    const int bimg = blockIdx.x >> 3;               // batch image
    const int row0 = (blockIdx.x & 7) * 4;          // first image row of slab

    // --- TDM: async-copy the padded weight matrix (64 x WROW f16) into LDS.
    //     D# per CDNA5 ISA ch.8: group0 = {flags, lds_addr, global_addr, type};
    //     group1 = {data_size, dims, tile dims, strides}. Issued by wave 0,
    //     overlapped with the activation tile loads below.
    if (wave == 0) {
        unsigned long long ga = (unsigned long long)Wt;
        unsigned int la = (unsigned int)(unsigned long long)(uintptr_t)wlds;
        u32x4 g0;
        g0[0] = 1u;                                    // count=1 (valid D#)
        g0[1] = la;                                    // lds_addr
        g0[2] = (unsigned int)ga;                      // global_addr[31:0]
        g0[3] = ((unsigned int)(ga >> 32) & 0x01FFFFFFu) | (2u << 30); // +type=2
        i32x8 g1;
        g1[0] = (int)(1u << 16);                       // data_size=1 (2 bytes)
        g1[1] = (int)((unsigned)WROW << 16);           // tensor_dim0 lo16
        g1[2] = (int)((unsigned)CH << 16);             // dim0 hi16=0 | dim1 lo16
        g1[3] = (int)((unsigned)WROW << 16);           // dim1 hi16=0 | tile_dim0
        g1[4] = (int)CH;                               // tile_dim1 | tile_dim2=0
        g1[5] = (int)WROW;                             // tensor_dim0_stride lo32
        g1[6] = (int)(((unsigned)(WROW * CH) & 0xFFFFu) << 16); // d1stride lo16
        g1[7] = (int)((unsigned)(WROW * CH) >> 16);    // d1stride hi
        i32x4 gz4 = {0, 0, 0, 0};
        i32x8 gz8 = {0, 0, 0, 0, 0, 0, 0, 0};
        __builtin_amdgcn_tensor_load_to_lds(g0, g1, gz4, gz4, gz8, 0);
    }

    // --- Cooperative, zero-padded channels-last tile load (x and h). ---------
    for (int i = tid; i < TROWS * TCOLS * 8; i += 256) {
        int seg = i & 7;                            // 8 f16 per segment
        int pix = i >> 3;
        int lr = pix / TCOLS;
        int lc = pix - lr * TCOLS;
        int gr = row0 - 1 + lr;
        int gc = lc - 1;
        uint4 vx = {0u, 0u, 0u, 0u};
        uint4 vh = {0u, 0u, 0u, 0u};
        if (gr >= 0 && gr < HH && gc >= 0 && gc < WW) {
            size_t g = ((size_t)(bimg * 1024 + gr * 32 + gc)) * 64 + seg * 8;
            vx = *(const uint4*)(xT + g);
            vh = *(const uint4*)(hin + g);
        }
        int l = pix * PIXP + seg * 8;
        *(uint4*)(&smem[l]) = vx;
        *(uint4*)(&smem[TILE_P + l]) = vh;
    }
    __builtin_amdgcn_s_wait_tensorcnt(0);   // wave0: TDM done before barrier
    __syncthreads();

    // A-fragment row for this lane (ISA 16-bit A 16x32 layout: both lane
    // halves cover M=0..15; lane-half selects the K sub-offset).
    const int mA    = lane & 15;
    const int mloc  = wave * 16 + mA;
    const int aprow = mloc >> 5;                    // 0..3 within slab
    const int apcol = mloc & 31;
    const int ncol  = lane & 15;                    // D/B column

    // Loop-invariant bases: one for A, one per N-tile for B. Every per-chunk
    // delta is a compile-time constant that fits the ds_load offset field,
    // so the unrolled loop body has no VALU address math (and thus no
    // WMMA->VALU hazard v_nops).
    const _Float16* ap  = smem + (aprow * TCOLS + apcol) * PIXP + hl * 16;
    const _Float16* wp0 = wlds + ncol * WROW + hl * 16;
    const _Float16* wp1 = wp0 + 16 * WROW;
    const _Float16* wp2 = wp0 + 32 * WROW;
    const _Float16* wp3 = wp0 + 48 * WROW;

    v8f acc0 = {}, acc1 = {}, acc2 = {}, acc3 = {};

#pragma unroll
    for (int kk = 0; kk < 36; ++kk) {
        const int toff = (kk < 18) ? 0 : TILE_P;    // x tile vs h tile
        const int kkl  = (kk < 18) ? kk : kk - 18;
        const int tap  = kkl >> 1;                  // (ky*3+kx) — constant
        const int c0   = (kkl & 1) << 5;            // channel sub-block 0/32
        const int ky   = tap / 3;
        const int kx   = tap - ky * 3;
        const int aoff = (ky * TCOLS + kx) * PIXP + c0 + toff;   // constant
        const int wo   = kk * 32;                                // constant

        v8h alo  = *(const v8h*)(ap + aoff);
        v8h ahi  = *(const v8h*)(ap + aoff + 8);
        v8h b0lo = *(const v8h*)(wp0 + wo);
        v8h b0hi = *(const v8h*)(wp0 + wo + 8);
        v8h b1lo = *(const v8h*)(wp1 + wo);
        v8h b1hi = *(const v8h*)(wp1 + wo + 8);
        v8h b2lo = *(const v8h*)(wp2 + wo);
        v8h b2hi = *(const v8h*)(wp2 + wo + 8);
        v8h b3lo = *(const v8h*)(wp3 + wo);
        v8h b3hi = *(const v8h*)(wp3 + wo + 8);

        v16h a = SHUF16(alo, ahi);
        acc0 = __builtin_amdgcn_wmma_f32_16x16x32_f16(
            false, a, false, SHUF16(b0lo, b0hi), (short)0, acc0, false, false);
        acc1 = __builtin_amdgcn_wmma_f32_16x16x32_f16(
            false, a, false, SHUF16(b1lo, b1hi), (short)0, acc1, false, false);
        acc2 = __builtin_amdgcn_wmma_f32_16x16x32_f16(
            false, a, false, SHUF16(b2lo, b2hi), (short)0, acc2, false, false);
        acc3 = __builtin_amdgcn_wmma_f32_16x16x32_f16(
            false, a, false, SHUF16(b3lo, b3hi), (short)0, acc3, false, false);
    }

    // Epilogue: D layout — lane l holds N = l&15; VGPR r holds M = 8*hl + r.
    v8f accv[4] = {acc0, acc1, acc2, acc3};
#pragma unroll
    for (int nt = 0; nt < 4; ++nt) {
        const int n = nt * 16 + ncol;
        const float bias = bx[n] + bh[n];
#pragma unroll
        for (int r = 0; r < 8; ++r) {
            const int M  = hl * 8 + r;
            const int ml = wave * 16 + M;
            const int pr = ml >> 5;
            const int pc = ml & 31;
            const int gp = bimg * 1024 + (row0 + pr) * 32 + pc;
            const size_t si = (size_t)gp * 64 + n;
            const float uo = u[si];
            const float ho = (uo > THRESH) ? 1.0f : 0.0f;   // spike at t-1
            const float un = DECAY * uo * (1.0f - ho) + accv[nt][r] + bias;
            const float sp = (un > THRESH) ? 1.0f : 0.0f;
            u[si] = un;
            hout[si] = (_Float16)sp;
            out[((size_t)((bimg * 64 + n) * 1024 + (row0 + pr) * 32 + pc)) * 20
                + t] = sp;
        }
    }
}

// ---------------------------------------------------------------------------
extern "C" void kernel_launch(void* const* d_in, const int* in_sizes, int n_in,
                              void* d_out, int out_size, void* d_ws,
                              size_t ws_size, hipStream_t stream) {
    const float* data = (const float*)d_in[0];
    const float* Wx   = (const float*)d_in[1];
    const float* bx   = (const float*)d_in[2];
    const float* Wh   = (const float*)d_in[3];
    const float* bh   = (const float*)d_in[4];
    float* out = (float*)d_out;

    // Workspace layout (all 16B aligned):
    //   xT  f16 [20][16*1024][64]     = 41,943,040 B
    //   Wt  f16 [64][1160] (padded)   =    148,480 B
    //   u   f32 [16*1024][64]         =  4,194,304 B
    //   h0  f16 [16*1024][64]         =  2,097,152 B
    //   h1  f16 [16*1024][64]         =  2,097,152 B
    char* ws = (char*)d_ws;
    const size_t nx     = (size_t)TT * BN * 1024 * 64;  // 20,971,520 elems
    const size_t nstate = (size_t)BN * 1024 * 64;       // 1,048,576 elems
    _Float16* xT = (_Float16*)ws;
    _Float16* Wt = (_Float16*)(ws + nx * 2);
    float*    u  = (float*)(ws + nx * 2 + (size_t)WLDS_HALFS * 2);
    _Float16* h0 = (_Float16*)((char*)u + nstate * 4);
    _Float16* h1 = h0 + nstate;

    // One-time relayout + f16 conversion (idempotent, deterministic).
    k_convert_x<<<(int)(nx / 256), 256, 0, stream>>>(data, xT, (int)nx);
    k_convert_w<<<(WLDS_HALFS + 255) / 256, 256, 0, stream>>>(Wx, Wh, Wt,
                                                              WLDS_HALFS);
    // Zero u + h0 + h1 (contiguous): 8,388,608 B = 524,288 uint4.
    const size_t zbytes = nstate * 4 + 2 * nstate * 2;
    k_zero16<<<(int)(zbytes / 16 / 256), 256, 0, stream>>>((uint4*)u,
                                                           (int)(zbytes / 16));

    // Serial scan over T; spike map double-buffered (blocks read neighbor
    // halos from hin while writing hout in the same launch).
    for (int t = 0; t < TT; ++t) {
        const _Float16* hin  = (t & 1) ? h1 : h0;
        _Float16*       hout = (t & 1) ? h0 : h1;
        k_lif_step<<<128, 256, DYN_LDS_BYTES, stream>>>(
            xT + (size_t)t * nstate, hin, Wt, bx, bh, u, hout, out, t);
    }
}
